// RelationalLayer_69389491634188
// MI455X (gfx1250) — compile-verified, matrix operand order checked by reading, not run on hardware
//
#include <hip/hip_runtime.h>

// ---------------------------------------------------------------------------
// RelationalLayer fused kernel for gfx1250 (MI455X), f16 WMMA path.
// ---------------------------------------------------------------------------

typedef _Float16 h8   __attribute__((ext_vector_type(8)));
typedef _Float16 v16h __attribute__((ext_vector_type(16)));
typedef float    v8f  __attribute__((ext_vector_type(8)));

#define NB   64      // batch
#define ND   64      // objects
#define NK   128     // feature dim
#define HID  256
#define NOUT 10

union FragU { v16h v; h8 h[2]; };

// Load a 16-half WMMA operand fragment: two contiguous 16-byte chunks
// (K = base..base+7 and K = base+16..base+23 for this lane), per the
// CDNA5 16-bit A/B operand VGPR layout.
__device__ __forceinline__ v16h load_frag(const _Float16* p) {
    FragU u;
    u.h[0] = *(const h8*)(p);
    u.h[1] = *(const h8*)(p + 16);
    return u.v;
}

// ---------------------------------------------------------------------------
// Prep kernels
// ---------------------------------------------------------------------------

// ti[b,j,:] = x[b,j,:] @ gW0[0:128,:],  tj[b,j,:] = x[b,j,:] @ gW0[128:256,:]
__global__ __launch_bounds__(256) void prep_ti_tj(const float* __restrict__ x,
                                                  const float* __restrict__ gW0,
                                                  float* __restrict__ ti,
                                                  float* __restrict__ tj) {
    __shared__ float xr[NK];
    const int r = blockIdx.x;        // b*ND + j
    const int n = threadIdx.x;       // 0..255
    if (n < NK) xr[n] = x[(size_t)r * NK + n];
    __syncthreads();
    float si = 0.f, sj = 0.f;
    #pragma unroll 4
    for (int k = 0; k < NK; ++k) {
        const float xv = xr[k];
        si += xv * gW0[k * HID + n];
        sj += xv * gW0[(NK + k) * HID + n];
    }
    ti[(size_t)r * HID + n] = si;
    tj[(size_t)r * HID + n] = sj;
}

// tq[b,:] = qst[b,:] @ gW0[256:384,:]; also zero x_g accumulator.
__global__ __launch_bounds__(256) void prep_tq_xg(const float* __restrict__ qst,
                                                  const float* __restrict__ gW0,
                                                  float* __restrict__ tq,
                                                  float* __restrict__ xg) {
    __shared__ float qr[NK];
    const int b = blockIdx.x;
    const int n = threadIdx.x;
    xg[b * HID + n] = 0.f;
    if (n < NK) qr[n] = qst[b * NK + n];
    __syncthreads();
    float s = 0.f;
    #pragma unroll 4
    for (int k = 0; k < NK; ++k) s += qr[k] * gW0[(2 * NK + k) * HID + n];
    tq[b * HID + n] = s;
}

// Transpose-convert gW1/gW2/gW3 (f32 [k][n]) -> Wt (f16 [n][k]) for B-fragment loads.
__global__ __launch_bounds__(256) void prep_wt(const float* __restrict__ gW1,
                                               const float* __restrict__ gW2,
                                               const float* __restrict__ gW3,
                                               _Float16* __restrict__ Wt) {
    const int id = blockIdx.x;          // 0 .. 3*HID-1
    const int L  = id / HID;
    const int nc = id % HID;
    const int k  = threadIdx.x;
    const float* W = (L == 0) ? gW1 : (L == 1) ? gW2 : gW3;
    Wt[(size_t)L * HID * HID + nc * HID + k] = (_Float16)W[k * HID + nc];
}

// ---------------------------------------------------------------------------
// Main fused kernel: one block per (b, i). 64 pair-rows x 256 cols in LDS,
// three fused 256x256 layers via v_wmma_f32_16x16x32_f16, reduce into x_g.
// ---------------------------------------------------------------------------
__global__ __launch_bounds__(256) void relnet_main(const float* __restrict__ ti,
                                                   const float* __restrict__ tj,
                                                   const float* __restrict__ tq,
                                                   const float* __restrict__ gb0,
                                                   const float* __restrict__ gb1,
                                                   const float* __restrict__ gb2,
                                                   const float* __restrict__ gb3,
                                                   const _Float16* __restrict__ Wt,
                                                   float* __restrict__ xg) {
    __shared__ __align__(16) _Float16 bufA[ND * HID];   // 32 KB
    __shared__ __align__(16) _Float16 bufB[ND * HID];   // 32 KB

    const int b   = blockIdx.x >> 6;
    const int i   = blockIdx.x & 63;
    const int tid = threadIdx.x;

    // ---- build h0[j,n] = relu(ti[b,j,n] + tj[b,i,n] + tq[b,n] + gb0[n]) ----
    {
        const int n = tid;
        const float base = tj[((size_t)b * ND + i) * HID + n] + tq[b * HID + n] + gb0[n];
        const float* tib = ti + (size_t)b * ND * HID + n;
        #pragma unroll 4
        for (int j = 0; j < ND; ++j) {
            const float v = tib[(size_t)j * HID] + base;
            bufA[j * HID + n] = (_Float16)fmaxf(v, 0.f);
        }
    }
    __syncthreads();

    const int wave  = tid >> 5;
    const int lane  = tid & 31;
    const int l15   = lane & 15;
    const int kbase = (lane < 16) ? 0 : 8;   // K sub-offset per lane half
    const int nb0   = wave * 32;             // this wave's column base (2 N-tiles)

    v8f acc[4][2];

    // GEMM over one 64x256 tile: out = in(64x256) @ W(256x256), W is [n][k].
    // All 4 A fragments are loaded into distinct registers before the WMMA
    // burst so the ds_load_b128s can be clause-grouped and overlapped.
    auto run_layer = [&](const _Float16* in, const _Float16* W) {
        const v8f zero = {};
        #pragma unroll
        for (int m = 0; m < 4; ++m) { acc[m][0] = zero; acc[m][1] = zero; }
        #pragma unroll
        for (int ks = 0; ks < 8; ++ks) {
            const int K0 = ks * 32;
            const v16h bf0 = load_frag(W + (nb0      + l15) * HID + K0 + kbase);
            const v16h bf1 = load_frag(W + (nb0 + 16 + l15) * HID + K0 + kbase);
            v16h af[4];
            #pragma unroll
            for (int m = 0; m < 4; ++m)
                af[m] = load_frag(in + (m * 16 + l15) * HID + K0 + kbase);
            #pragma unroll
            for (int m = 0; m < 4; ++m) {
                acc[m][0] = __builtin_amdgcn_wmma_f32_16x16x32_f16(
                    false, af[m], false, bf0, (short)0, acc[m][0], false, false);
                acc[m][1] = __builtin_amdgcn_wmma_f32_16x16x32_f16(
                    false, af[m], false, bf1, (short)0, acc[m][1], false, false);
            }
        }
    };

    // bias + relu + store f16 to the other buffer
    auto epilogue_store = [&](const float* bias, _Float16* out) {
        #pragma unroll
        for (int m = 0; m < 4; ++m) {
            #pragma unroll
            for (int nt = 0; nt < 2; ++nt) {
                const int col = nb0 + nt * 16 + l15;
                const float bb = bias[col];
                #pragma unroll
                for (int r = 0; r < 8; ++r) {
                    const int row = m * 16 + ((lane < 16) ? r : r + 8);
                    const float v = acc[m][nt][r] + bb;
                    out[row * HID + col] = (_Float16)fmaxf(v, 0.f);
                }
            }
        }
    };

    // bias + relu + column-sum over the 64 rows, atomic-accumulate into x_g[b,:]
    auto epilogue_reduce = [&](const float* bias, float* xgrow) {
        #pragma unroll
        for (int nt = 0; nt < 2; ++nt) {
            const int col = nb0 + nt * 16 + l15;
            const float bb = bias[col];
            float s = 0.f;
            #pragma unroll
            for (int m = 0; m < 4; ++m)
                #pragma unroll
                for (int r = 0; r < 8; ++r)
                    s += fmaxf(acc[m][nt][r] + bb, 0.f);
            // lanes l and l+16 hold complementary row halves of the same column
            s += __shfl_down(s, 16, 32);
            if (lane < 16) atomicAdd(&xgrow[col], s);
        }
    };

    run_layer(bufA, Wt);                       // layer 1: A -> B
    epilogue_store(gb1, bufB);
    __syncthreads();
    run_layer(bufB, Wt + HID * HID);           // layer 2: B -> A
    epilogue_store(gb2, bufA);
    __syncthreads();
    run_layer(bufA, Wt + 2 * HID * HID);       // layer 3: A -> reduce
    epilogue_reduce(gb3, xg + b * HID);
}

// ---------------------------------------------------------------------------
// Final tiny MLP: x_g -> relu(fW1) -> relu(fW2) -> fW3. One block per b.
// ---------------------------------------------------------------------------
__global__ __launch_bounds__(256) void final_mlp(const float* __restrict__ xg,
                                                 const float* __restrict__ fW1,
                                                 const float* __restrict__ fb1,
                                                 const float* __restrict__ fW2,
                                                 const float* __restrict__ fb2,
                                                 const float* __restrict__ fW3,
                                                 const float* __restrict__ fb3,
                                                 float* __restrict__ out) {
    __shared__ float h0[HID], h1[HID], h2[HID];
    const int b = blockIdx.x;
    const int t = threadIdx.x;
    h0[t] = xg[b * HID + t];
    __syncthreads();
    float s = 0.f;
    #pragma unroll 4
    for (int k = 0; k < HID; ++k) s += h0[k] * fW1[k * HID + t];
    h1[t] = fmaxf(s + fb1[t], 0.f);
    __syncthreads();
    s = 0.f;
    #pragma unroll 4
    for (int k = 0; k < HID; ++k) s += h1[k] * fW2[k * HID + t];
    h2[t] = fmaxf(s + fb2[t], 0.f);
    __syncthreads();
    if (t < NOUT) {
        s = 0.f;
        #pragma unroll 4
        for (int k = 0; k < HID; ++k) s += h2[k] * fW3[k * NOUT + t];
        out[b * NOUT + t] = s + fb3[t];
    }
}

// ---------------------------------------------------------------------------
// Launch
// ---------------------------------------------------------------------------
extern "C" void kernel_launch(void* const* d_in, const int* in_sizes, int n_in,
                              void* d_out, int out_size, void* d_ws, size_t ws_size,
                              hipStream_t stream) {
    const float* x   = (const float*)d_in[0];
    const float* qst = (const float*)d_in[1];
    const float* gW0 = (const float*)d_in[2];
    const float* gb0 = (const float*)d_in[3];
    const float* gW1 = (const float*)d_in[4];
    const float* gb1 = (const float*)d_in[5];
    const float* gW2 = (const float*)d_in[6];
    const float* gb2 = (const float*)d_in[7];
    const float* gW3 = (const float*)d_in[8];
    const float* gb3 = (const float*)d_in[9];
    const float* fW1 = (const float*)d_in[10];
    const float* fb1 = (const float*)d_in[11];
    const float* fW2 = (const float*)d_in[12];
    const float* fb2 = (const float*)d_in[13];
    const float* fW3 = (const float*)d_in[14];
    const float* fb3 = (const float*)d_in[15];
    float* out = (float*)d_out;

    // workspace layout (f32-element offsets; all 16B-aligned)
    char* ws = (char*)d_ws;
    const size_t TI_E = (size_t)NB * ND * HID;     // 1,048,576
    float*     ti = (float*)ws;
    float*     tj = ti + TI_E;
    float*     tq = tj + TI_E;                     // NB*HID
    float*     xg = tq + (size_t)NB * HID;         // NB*HID
    _Float16*  Wt = (_Float16*)(xg + (size_t)NB * HID);  // 3*HID*HID halves

    prep_ti_tj<<<NB * ND, 256, 0, stream>>>(x, gW0, ti, tj);
    prep_tq_xg<<<NB, 256, 0, stream>>>(qst, gW0, tq, xg);
    prep_wt<<<3 * HID, 256, 0, stream>>>(gW1, gW2, gW3, Wt);
    relnet_main<<<NB * ND, 256, 0, stream>>>(ti, tj, tq, gb0, gb1, gb2, gb3, Wt, xg);
    final_mlp<<<NB, 256, 0, stream>>>(xg, fW1, fb1, fW2, fb2, fW3, fb3, out);
}
